// GCN_Net_38139309588567
// MI455X (gfx1250) — matrix operand order repaired and checked by reading, not compiled, not run on hardware
//
#include <hip/hip_runtime.h>
#include <math.h>

#define NNODES 50000

typedef __attribute__((ext_vector_type(16))) __bf16 v16bf;
typedef __attribute__((ext_vector_type(8)))  __bf16 v8bf;
typedef __attribute__((ext_vector_type(4)))  __bf16 v4bf;
typedef __attribute__((ext_vector_type(8)))  float  v8f;

// ---------------------------------------------------------------- degree prep
__global__ void k_init_deg(float* __restrict__ deg, int n) {
    int i = blockIdx.x * blockDim.x + threadIdx.x;
    if (i < n) deg[i] = 1.0f;               // self loop contributes 1
}

__global__ void k_count_deg(const int* __restrict__ col, float* __restrict__ deg, int E) {
    int i = blockIdx.x * blockDim.x + threadIdx.x;
    if (i < E) atomicAdd(&deg[col[i]], 1.0f);
}

__global__ void k_rsqrt_inplace(float* __restrict__ deg, int n) {
    int i = blockIdx.x * blockDim.x + threadIdx.x;
    if (i < n) deg[i] = rsqrtf(deg[i]);     // deg >= 1 always (self loop)
}

// ---------------------------------------------------------------- bf16 WMMA GEMM
// C[M,N] = A[M,K] (f32) * B[K,N] (f32), bf16 inputs / f32 accumulate.
// Block tile 128x64, 256 threads = 8 waves (4 M x 2 N); each wave owns a 32x32
// macro-tile = 2x2 WMMA tiles. Double-buffered LDS, one barrier per K-step;
// next-tile global loads are issued before the WMMAs to hide latency.
// NOTE: requires K % 4 == 0 and N % 4 == 0 (holds: K in {500,256}, N in {256,40});
// float4 chunks are then all-valid or all-invalid -> branchless clamp+select.
constexpr int BM = 128, BN = 64, BK = 32;
constexpr int LDA = 40;                 // 32 + 8 pad -> 80B rows (16B aligned)
constexpr int LDB = 40;

__device__ __forceinline__ void glb_load_A(const float* __restrict__ A, int M, int K,
                                           int m0, int kb, int tid, float4 (&ra)[4]) {
#pragma unroll
    for (int j = 0; j < 4; ++j) {
        int idx = tid + j * 256;                 // 0..1023
        int r   = idx >> 3;                      // row 0..127
        int kq  = (idx & 7) << 2;                // 0,4,..,28
        int gm  = m0 + r, gk = kb + kq;
        bool vr = gm < M;
        bool vk = gk + 3 < K;
        const float* p = A + (long)(vr ? gm : 0) * K + (vk ? gk : 0);
        float4 v = *(const float4*)p;            // always in-bounds address
        if (!(vr && vk)) v = make_float4(0.f, 0.f, 0.f, 0.f);
        ra[j] = v;
    }
}

__device__ __forceinline__ void glb_load_B(const float* __restrict__ B, int N, int K,
                                           int n0, int kb, int tid, float4 (&rb)[2]) {
#pragma unroll
    for (int j = 0; j < 2; ++j) {
        int idx = tid + j * 256;                 // 0..511
        int kk  = idx >> 4;                      // K row 0..31
        int nq  = (idx & 15) << 2;               // 0,4,..,60
        int gk  = kb + kk, gn = n0 + nq;
        bool vk = gk < K;
        bool vn = gn + 3 < N;
        const float* p = B + (long)(vk ? gk : 0) * N + (vn ? gn : 0);
        float4 v = *(const float4*)p;
        if (!(vk && vn)) v = make_float4(0.f, 0.f, 0.f, 0.f);
        rb[j] = v;
    }
}

__device__ __forceinline__ void stage_A(__bf16* __restrict__ As, int tid,
                                        const float4 (&ra)[4]) {
#pragma unroll
    for (int j = 0; j < 4; ++j) {
        int idx = tid + j * 256;
        int r   = idx >> 3;
        int kq  = (idx & 7) << 2;
        v4bf pk;
        pk[0] = (__bf16)ra[j].x; pk[1] = (__bf16)ra[j].y;
        pk[2] = (__bf16)ra[j].z; pk[3] = (__bf16)ra[j].w;
        *(v4bf*)(&As[r * LDA + kq]) = pk;        // 8B ds_store
    }
}

__device__ __forceinline__ void stage_B(__bf16* __restrict__ Bs, int tid,
                                        const float4 (&rb)[2]) {
#pragma unroll
    for (int j = 0; j < 2; ++j) {
        int idx = tid + j * 256;
        int kk  = idx >> 4;
        int nq  = (idx & 15) << 2;
        Bs[(nq + 0) * LDB + kk] = (__bf16)rb[j].x;   // transposed: Bs[n][k]
        Bs[(nq + 1) * LDB + kk] = (__bf16)rb[j].y;
        Bs[(nq + 2) * LDB + kk] = (__bf16)rb[j].z;
        Bs[(nq + 3) * LDB + kk] = (__bf16)rb[j].w;
    }
}

__global__ __launch_bounds__(256)
void k_gemm_bf16(const float* __restrict__ A, const float* __restrict__ B,
                 float* __restrict__ C, int M, int N, int K) {
    __shared__ __align__(16) __bf16 Asd[2][BM * LDA];   // [M][K], double buffered
    __shared__ __align__(16) __bf16 Bsd[2][BN * LDB];   // [N][K] transposed

    const int tid   = threadIdx.x;
    const int lane  = tid & 31;
    const int wid   = tid >> 5;
    const int wm    = wid & 3;              // M macro-tile 0..3 (32 rows)
    const int wn    = wid >> 2;             // N macro-tile 0..1 (32 cols)
    const int l16   = lane & 15;
    const int khalf = lane >> 4;

    const int m0 = blockIdx.x * BM;
    const int n0 = blockIdx.y * BN;

    v8f acc00 = {}, acc01 = {}, acc10 = {}, acc11 = {};

    float4 ra[4], rb[2];
    glb_load_A(A, M, K, m0, 0, tid, ra);
    glb_load_B(B, N, K, n0, 0, tid, rb);
    stage_A(Asd[0], tid, ra);
    stage_B(Bsd[0], tid, rb);
    __syncthreads();

    const int ktiles = (K + BK - 1) / BK;
    for (int kt = 0; kt < ktiles; ++kt) {
        const int cur = kt & 1, nxt = cur ^ 1;
        const bool more = (kt + 1) < ktiles;
        if (more) {                         // issue next-tile global loads early
            glb_load_A(A, M, K, m0, (kt + 1) * BK, tid, ra);
            glb_load_B(B, N, K, n0, (kt + 1) * BK, tid, rb);
        }

        // ---- fragments from current buffer
        const __bf16* Ac = Asd[cur];
        const __bf16* Bc = Bsd[cur];
        v16bf a[2], b[2];
#pragma unroll
        for (int mt = 0; mt < 2; ++mt) {
            const __bf16* ap = &Ac[(wm * 32 + mt * 16 + l16) * LDA];
            v8bf lo = *(const v8bf*)(ap + 8 * khalf);
            v8bf hi = *(const v8bf*)(ap + 16 + 8 * khalf);
#pragma unroll
            for (int i = 0; i < 8; ++i) { a[mt][i] = lo[i]; a[mt][8 + i] = hi[i]; }
        }
#pragma unroll
        for (int nt = 0; nt < 2; ++nt) {
            const __bf16* bp = &Bc[(wn * 32 + nt * 16 + l16) * LDB];
            v8bf lo = *(const v8bf*)(bp + 8 * khalf);
            v8bf hi = *(const v8bf*)(bp + 16 + 8 * khalf);
#pragma unroll
            for (int i = 0; i < 8; ++i) { b[nt][i] = lo[i]; b[nt][8 + i] = hi[i]; }
        }

        acc00 = __builtin_amdgcn_wmma_f32_16x16x32_bf16(false, a[0], false, b[0],
                                                        (short)0, acc00, false, false);
        acc01 = __builtin_amdgcn_wmma_f32_16x16x32_bf16(false, a[0], false, b[1],
                                                        (short)0, acc01, false, false);
        acc11 = __builtin_amdgcn_wmma_f32_16x16x32_bf16(false, a[1], false, b[1],
                                                        (short)0, acc11, false, false);
        acc10 = __builtin_amdgcn_wmma_f32_16x16x32_bf16(false, a[1], false, b[0],
                                                        (short)0, acc10, false, false);

        if (more) {                         // convert + store prefetched tile
            stage_A(Asd[nxt], tid, ra);
            stage_B(Bsd[nxt], tid, rb);
        }
        __syncthreads();                    // single barrier per K-step
    }

    // ---- store D: VGPR r -> M = r + 8*khalf, N = l16
    v8f* accs[2][2] = { { &acc00, &acc01 }, { &acc10, &acc11 } };
#pragma unroll
    for (int mt = 0; mt < 2; ++mt) {
#pragma unroll
        for (int nt = 0; nt < 2; ++nt) {
            int gn = n0 + wn * 32 + nt * 16 + l16;
            if (gn >= N) continue;
            v8f& c = *accs[mt][nt];
#pragma unroll
            for (int r = 0; r < 8; ++r) {
                int gm = m0 + wm * 32 + mt * 16 + khalf * 8 + r;
                if (gm < M) C[(long)gm * N + gn] = c[r];
            }
        }
    }
}

// ---------------------------------------------------------------- SpMM scatter
// dst[col[e], :] += dinv[row[e]]*dinv[col[e]] * src[row[e], :]
template <int F>
__global__ void k_spmm(const float* __restrict__ src, float* __restrict__ dst,
                       const int* __restrict__ row, const int* __restrict__ col,
                       const float* __restrict__ dinv, int E) {
    constexpr int CH = F >> 2;
    long t      = (long)blockIdx.x * blockDim.x + threadIdx.x;
    long total  = (long)E * CH;
    long stride = (long)gridDim.x * blockDim.x;
    for (; t < total; t += stride) {
        int e  = (int)(t / CH);
        int fc = (int)(t % CH) << 2;
        int r = row[e], c = col[e];
        float w = dinv[r] * dinv[c];
        float4 v = *(const float4*)(src + (long)r * F + fc);
        float* d = dst + (long)c * F + fc;
        atomicAdd(d + 0, v.x * w);
        atomicAdd(d + 1, v.y * w);
        atomicAdd(d + 2, v.z * w);
        atomicAdd(d + 3, v.w * w);
    }
}

// ---------------------------------------------------------------- epilogues
template <int F>
__global__ void k_self_bias_elu(const float* __restrict__ xl, float* __restrict__ h,
                                const float* __restrict__ dinv, const float* __restrict__ b,
                                int n) {
    long t = (long)blockIdx.x * blockDim.x + threadIdx.x;
    if (t >= (long)n * F) return;
    int i = (int)(t / F), f = (int)(t % F);
    float di = dinv[i];
    float v = h[t] + di * di * xl[t] + b[f];
    h[t] = (v > 0.0f) ? v : (expf(v) - 1.0f);
}

template <int F>
__global__ void k_self_bias_acc(const float* __restrict__ h2, float* __restrict__ acc,
                                const float* __restrict__ dinv, const float* __restrict__ b,
                                int n) {
    long t = (long)blockIdx.x * blockDim.x + threadIdx.x;
    if (t >= (long)n * F) return;
    int i = (int)(t / F), f = (int)(t % F);
    float di = dinv[i];
    acc[t] += di * di * h2[t] + b[f];
}

// ---------------------------------------------------------------- log_softmax
template <int F>
__global__ void k_logsoftmax(const float* __restrict__ acc, float* __restrict__ out, int n) {
    int i = blockIdx.x * blockDim.x + threadIdx.x;
    if (i >= n) return;
    const float* v = acc + (long)i * F;
    float m = -INFINITY;
    for (int f = 0; f < F; ++f) m = fmaxf(m, v[f]);
    float s = 0.0f;
    for (int f = 0; f < F; ++f) s += expf(v[f] - m);
    float ls = logf(s);
    float* o = out + (long)i * F;
    for (int f = 0; f < F; ++f) o[f] = v[f] - m - ls;
}

// ----------------------------------------------------------------------------
static inline size_t align_up(size_t x) { return (x + 255) & ~(size_t)255; }

extern "C" void kernel_launch(void* const* d_in, const int* in_sizes, int n_in,
                              void* d_out, int out_size, void* d_ws, size_t ws_size,
                              hipStream_t stream) {
    const float* x  = (const float*)d_in[0];
    const int*   ei = (const int*)d_in[1];
    const float* W1[2] = { (const float*)d_in[2], (const float*)d_in[6] };
    const float* b1[2] = { (const float*)d_in[3], (const float*)d_in[7] };
    const float* W2[2] = { (const float*)d_in[4], (const float*)d_in[8] };
    const float* b2[2] = { (const float*)d_in[5], (const float*)d_in[9] };

    const int N   = NNODES;
    const int E   = in_sizes[1] / 2;
    const int Fin = in_sizes[0] / N;      // 500
    const int H   = 256;
    const int C   = 40;
    const int* row = ei;
    const int* col = ei + E;

    // workspace (~110.6 MB): dinv | xl | h | acc ; h2 aliases xl
    char* w = (char*)d_ws;
    size_t off = 0;
    float* dinv = (float*)(w + off); off += align_up((size_t)N * sizeof(float));
    float* xl   = (float*)(w + off); off += align_up((size_t)N * H * sizeof(float));
    float* h    = (float*)(w + off); off += align_up((size_t)N * H * sizeof(float));
    float* acc  = (float*)(w + off); off += align_up((size_t)N * C * sizeof(float));
    float* h2   = xl;                     // xl dead after the ELU epilogue
    (void)ws_size; (void)n_in; (void)out_size;

    // ---- normalization: deg = 1 + count(col); dinv = deg^-1/2
    k_init_deg<<<(N + 255) / 256, 256, 0, stream>>>(dinv, N);
    k_count_deg<<<(E + 255) / 256, 256, 0, stream>>>(col, dinv, E);
    k_rsqrt_inplace<<<(N + 255) / 256, 256, 0, stream>>>(dinv, N);

    hipMemsetAsync(acc, 0, (size_t)N * C * sizeof(float), stream);

    for (int br = 0; br < 2; ++br) {
        // ---- layer 1: xl = x @ W1
        dim3 g1((N + BM - 1) / BM, (H + BN - 1) / BN);
        k_gemm_bf16<<<g1, 256, 0, stream>>>(x, W1[br], xl, N, H, Fin);
        hipMemsetAsync(h, 0, (size_t)N * H * sizeof(float), stream);
        {
            long total = (long)E * (H >> 2);
            k_spmm<256><<<(int)((total + 255) / 256), 256, 0, stream>>>(
                xl, h, row, col, dinv, E);
        }
        {
            long total = (long)N * H;
            k_self_bias_elu<256><<<(int)((total + 255) / 256), 256, 0, stream>>>(
                xl, h, dinv, b1[br], N);
        }

        // ---- layer 2: h2 = h @ W2
        dim3 g2((N + BM - 1) / BM, (C + BN - 1) / BN);
        k_gemm_bf16<<<g2, 256, 0, stream>>>(h, W2[br], h2, N, C, H);
        {
            long total = (long)E * (C >> 2);
            k_spmm<40><<<(int)((total + 255) / 256), 256, 0, stream>>>(
                h2, acc, row, col, dinv, E);
        }
        {
            long total = (long)N * C;
            k_self_bias_acc<40><<<(int)((total + 255) / 256), 256, 0, stream>>>(
                h2, acc, dinv, b2[br], N);
        }
    }

    k_logsoftmax<40><<<(N + 255) / 256, 256, 0, stream>>>(acc, (float*)d_out, N);
}
